// HilbertDilatedAttention_58926951301489
// MI455X (gfx1250) — matrix operand order repaired and checked by reading, not compiled, not run on hardware
//
#include <hip/hip_runtime.h>

// ---------------------------------------------------------------------------
// Types for CDNA5 WMMA
// ---------------------------------------------------------------------------
typedef __bf16 v16bf __attribute__((ext_vector_type(16)));
typedef __bf16 v8bf  __attribute__((ext_vector_type(8)));
typedef __bf16 v4bf  __attribute__((ext_vector_type(4)));
typedef float  v8f   __attribute__((ext_vector_type(8)));
typedef int    v4i_gcc __attribute__((vector_size(16)));   // matches builtin param

#define WMMA_BF16(a, b, c) \
  __builtin_amdgcn_wmma_f32_16x16x32_bf16(false, (a), false, (b), (short)0, (c), false, false)

// ---------------------------------------------------------------------------
// CDNA5 async global->LDS copy (GLOBAL_LOAD_ASYNC_TO_LDS_B128, ASYNCcnt) with
// a synchronous fallback if the builtin is absent on this toolchain.
// Signature (from hipcc diagnostic): (v4i __device__*, v4i __shared__*, Imm, Imm)
// ---------------------------------------------------------------------------
#if defined(__AMDGCN__) && __has_builtin(__builtin_amdgcn_global_load_async_to_lds_b128) && \
    __has_builtin(__builtin_amdgcn_s_wait_asynccnt)
#define HAVE_ASYNC_LDS 1
#else
#define HAVE_ASYNC_LDS 0
#endif

__device__ __forceinline__ void async_copy_b128(const __bf16* g, __bf16* l) {
#if HAVE_ASYNC_LDS
  __builtin_amdgcn_global_load_async_to_lds_b128((v4i_gcc*)g, (v4i_gcc*)l, 0, 0);
#else
  *(v8bf*)l = *(const v8bf*)g;
#endif
}

#if HAVE_ASYNC_LDS
#define WAIT_ASYNC(n) __builtin_amdgcn_s_wait_asynccnt(n)
#else
#define WAIT_ASYNC(n) ((void)0)
#endif

// ---------------------------------------------------------------------------
// Problem constants (fixed by the reference harness)
// ---------------------------------------------------------------------------
static constexpr int kB    = 2;
static constexpr int kS    = 8192;
static constexpr int kD    = 1024;   // model dim
static constexpr int kH    = 16;
static constexpr int kDh   = 64;
static constexpr int kSeg  = 128;
static constexpr int kNSeg = kS / kSeg;      // 64
static constexpr int kM    = kB * kS;        // 16384 rows
static constexpr int kNqkv = 3 * kD;         // 3072
static constexpr int kG    = 91;             // ceil(sqrt(8192))

// ---------------------------------------------------------------------------
// Kernel 0: boustrophedon ("Hilbert") permutation map, closed form.
// ---------------------------------------------------------------------------
__global__ void hilbert_map_kernel(int* __restrict__ m, int S, int g) {
  int lp = blockIdx.x * blockDim.x + threadIdx.x;
  if (lp >= S) return;
  int r = lp / g, c = lp % g;
  int limit = S - r * g; if (limit > g) limit = g;
  int cum = r * g;       if (cum > S)   cum = S;
  int before = (r & 1) ? (limit - 1 - c) : c;
  m[lp] = cum + before;
}

// ---------------------------------------------------------------------------
// Kernel 1: f32 -> bf16 elementwise (vectorized x4)
// ---------------------------------------------------------------------------
__global__ void cvt_f32_bf16_kernel(const float* __restrict__ x,
                                    __bf16* __restrict__ y, int n) {
  int i = (blockIdx.x * blockDim.x + threadIdx.x) * 4;
  if (i >= n) return;
  float4 v = *(const float4*)(x + i);
  v4bf o;
  o.x = (__bf16)v.x; o.y = (__bf16)v.y; o.z = (__bf16)v.z; o.w = (__bf16)v.w;
  *(v4bf*)(y + i) = o;
}

// ---------------------------------------------------------------------------
// Kernel 2: W[K][N] f32 -> Wt[N][K] bf16 (LDS-tiled transpose, 32x32 tiles)
// ---------------------------------------------------------------------------
__global__ void transpose_cvt_kernel(const float* __restrict__ W,
                                     __bf16* __restrict__ Wt, int K, int N) {
  __shared__ float tile[32][33];
  int n0 = blockIdx.x * 32;
  int k0 = blockIdx.y * 32;
  for (int i = threadIdx.y; i < 32; i += 8)
    tile[i][threadIdx.x] = W[(size_t)(k0 + i) * N + n0 + threadIdx.x];
  __syncthreads();
  for (int i = threadIdx.y; i < 32; i += 8)
    Wt[(size_t)(n0 + i) * K + k0 + threadIdx.x] = (__bf16)tile[threadIdx.x][i];
}

// ---------------------------------------------------------------------------
// Kernel 3: bf16 WMMA GEMM   C[M][N] = A[M][K] * Bt[N][K]^T
// 256 threads = 8 waves, block tile 128(M) x 256(N), wave tile 64x64.
// K staged 32-wide into LDS with double-buffered async copies:
//   stage(next); s_wait_asynccnt 6 (leave the 6 prefetch ops in flight);
//   barrier; 16 x v_wmma per wave from ds_load_b128 fragments.
// ---------------------------------------------------------------------------
template <typename OutT>
__global__ void __launch_bounds__(256)
gemm_bf16_kernel(const __bf16* __restrict__ A, const __bf16* __restrict__ Bt,
                 OutT* __restrict__ C, int M, int N, int K) {
  __shared__ __bf16 Ab[2][128 * 32];   // 8 KB each
  __shared__ __bf16 Bb[2][256 * 32];   // 16 KB each

  int tid  = threadIdx.x;
  int lane = tid & 31;
  int w    = tid >> 5;
  int lo = lane & 15, hi = lane >> 4;
  int bm = blockIdx.y * 128;
  int bn = blockIdx.x * 256;
  int wmL = (w >> 2) * 64;   // wave tile origin inside block (M)
  int wnL = (w & 3) * 64;    // wave tile origin inside block (N)

  // Stage one 32-wide K slice of the A (128x32) and B (256x32) tiles.
  // Per thread: 2 A-chunks + 4 B-chunks of 16B => 6 async ops per wave.
  auto stage = [&](int buf, int k0) {
#pragma unroll
    for (int it = 0; it < 2; ++it) {
      int c = tid + it * 256;            // 0..511 : c = r*4 + q
      int r = c >> 2, q = c & 3;
      async_copy_b128(A + (size_t)(bm + r) * K + k0 + q * 8,
                      &Ab[buf][r * 32 + q * 8]);
    }
#pragma unroll
    for (int it = 0; it < 4; ++it) {
      int c = tid + it * 256;            // 0..1023
      int r = c >> 2, q = c & 3;
      async_copy_b128(Bt + (size_t)(bn + r) * K + k0 + q * 8,
                      &Bb[buf][r * 32 + q * 8]);
    }
  };

  v8f zero = {0.f, 0.f, 0.f, 0.f, 0.f, 0.f, 0.f, 0.f};
  v8f acc[4][4];
#pragma unroll
  for (int mi = 0; mi < 4; ++mi)
#pragma unroll
    for (int ni = 0; ni < 4; ++ni) acc[mi][ni] = zero;

  stage(0, 0);
  for (int k0 = 0; k0 < K; k0 += 32) {
    int buf = (k0 >> 5) & 1;
    if (k0 + 32 < K) {
      stage(buf ^ 1, k0 + 32);
      WAIT_ASYNC(6);                 // prior tile complete; prefetch in flight
    } else {
      WAIT_ASYNC(0);
    }
    __syncthreads();

    v16bf af[4], bfr[4];
#pragma unroll
    for (int mi = 0; mi < 4; ++mi)
      af[mi] = *(const v16bf*)(&Ab[buf][(wmL + mi * 16 + lo) * 32 + hi * 16]);
#pragma unroll
    for (int ni = 0; ni < 4; ++ni)
      bfr[ni] = *(const v16bf*)(&Bb[buf][(wnL + ni * 16 + lo) * 32 + hi * 16]);
#pragma unroll
    for (int mi = 0; mi < 4; ++mi)
#pragma unroll
      for (int ni = 0; ni < 4; ++ni)
        acc[mi][ni] = WMMA_BF16(af[mi], bfr[ni], acc[mi][ni]);
    __syncthreads();                 // protect buf from next stage overwrite
  }

  // C layout: lane 0-15 -> M = v, lane 16-31 -> M = v+8 ; N = lane&15
#pragma unroll
  for (int mi = 0; mi < 4; ++mi)
#pragma unroll
    for (int ni = 0; ni < 4; ++ni)
#pragma unroll
      for (int v = 0; v < 8; ++v)
        C[(size_t)(bm + wmL + mi * 16 + v + 8 * hi) * N + bn + wnL + ni * 16 + lo] =
            (OutT)acc[mi][ni][v];
}

// ---------------------------------------------------------------------------
// Kernel 4: segmented dilated attention with Hilbert gather/scatter.
// Block = (seg, head, batch), 128 threads = 4 waves; each wave: 32 query rows.
// ---------------------------------------------------------------------------
__global__ void __launch_bounds__(128)
attn_kernel(const __bf16* __restrict__ qkv, const int* __restrict__ mmap,
            __bf16* __restrict__ attn_out) {
  __shared__ __bf16 Ks[64 * 64];        // K' row-major [key][d]
  __shared__ __bf16 VTs[64 * 64];       // V'^T [d][key]
  __shared__ __bf16 Ps[4 * 32 * 64];    // per-wave softmax probs

  int n = blockIdx.x, h = blockIdx.y, b = blockIdx.z;
  int tid = threadIdx.x;
  int lane = tid & 31, w = tid >> 5;
  int lo = lane & 15, hi = lane >> 4;

  // ---- stage K' via async copies, V' transposed via ds stores (gather via m)
  {
    int t = tid >> 1, half = tid & 1;              // key row, 32-col half
    int orig = mmap[n * kSeg + 2 * t];             // dilation 2
    const __bf16* base = qkv + (size_t)(b * kS + orig) * kNqkv + h * kDh;
    const __bf16* krow = base + kD + half * 32;
    async_copy_b128(krow,     Ks + t * 64 + half * 32);
    async_copy_b128(krow + 8, Ks + t * 64 + half * 32 + 8);
    async_copy_b128(krow + 16, Ks + t * 64 + half * 32 + 16);
    async_copy_b128(krow + 24, Ks + t * 64 + half * 32 + 24);
    const __bf16* vrow = base + 2 * kD + half * 32;
#pragma unroll
    for (int d2 = 0; d2 < 32; ++d2)
      VTs[(half * 32 + d2) * 64 + t] = vrow[d2];
  }
  WAIT_ASYNC(0);
  __syncthreads();

  // ---- Q fragments (gathered straight from global via m)
  v16bf qf[2][2];
#pragma unroll
  for (int mi = 0; mi < 2; ++mi)
#pragma unroll
    for (int ks = 0; ks < 2; ++ks) {
      int orig = mmap[n * kSeg + w * 32 + mi * 16 + lo];
      qf[mi][ks] = *(const v16bf*)(qkv + (size_t)(b * kS + orig) * kNqkv +
                                   h * kDh + ks * 32 + hi * 16);
    }

  // ---- scores = Q * K'^T
  v8f zero = {0.f, 0.f, 0.f, 0.f, 0.f, 0.f, 0.f, 0.f};
  v8f sacc[2][4];
#pragma unroll
  for (int mi = 0; mi < 2; ++mi)
#pragma unroll
    for (int ni = 0; ni < 4; ++ni) sacc[mi][ni] = zero;
#pragma unroll
  for (int ks = 0; ks < 2; ++ks) {
    v16bf kf[4];
#pragma unroll
    for (int ni = 0; ni < 4; ++ni)
      kf[ni] = *(const v16bf*)(Ks + (ni * 16 + lo) * 64 + ks * 32 + hi * 16);
#pragma unroll
    for (int mi = 0; mi < 2; ++mi)
#pragma unroll
      for (int ni = 0; ni < 4; ++ni)
        sacc[mi][ni] = WMMA_BF16(qf[mi][ks], kf[ni], sacc[mi][ni]);
  }

  // ---- softmax over 64 keys (4 acc regs per row-slot + 16-lane reduction)
  const float scale = 0.125f;  // 1/sqrt(64)
#pragma unroll
  for (int mi = 0; mi < 2; ++mi) {
#pragma unroll
    for (int v = 0; v < 8; ++v) {
      float a0 = sacc[mi][0][v] * scale, a1 = sacc[mi][1][v] * scale;
      float a2 = sacc[mi][2][v] * scale, a3 = sacc[mi][3][v] * scale;
      float mx = fmaxf(fmaxf(a0, a1), fmaxf(a2, a3));
#pragma unroll
      for (int off = 1; off < 16; off <<= 1)
        mx = fmaxf(mx, __shfl_xor(mx, off, 32));
      float e0 = __expf(a0 - mx), e1 = __expf(a1 - mx);
      float e2 = __expf(a2 - mx), e3 = __expf(a3 - mx);
      float ssum = e0 + e1 + e2 + e3;
#pragma unroll
      for (int off = 1; off < 16; off <<= 1)
        ssum += __shfl_xor(ssum, off, 32);
      float rinv = 1.0f / ssum;
      int rloc = mi * 16 + v + 8 * hi;                  // 0..31 within wave
      __bf16* pr = Ps + w * 2048 + rloc * 64 + lo;
      pr[0]  = (__bf16)(e0 * rinv);
      pr[16] = (__bf16)(e1 * rinv);
      pr[32] = (__bf16)(e2 * rinv);
      pr[48] = (__bf16)(e3 * rinv);
    }
  }
  __syncthreads();

  // ---- out = P * V'
  v8f oacc[2][4];
#pragma unroll
  for (int mi = 0; mi < 2; ++mi)
#pragma unroll
    for (int ni = 0; ni < 4; ++ni) oacc[mi][ni] = zero;
#pragma unroll
  for (int ks = 0; ks < 2; ++ks) {
    v16bf vf[4], pf[2];
#pragma unroll
    for (int ni = 0; ni < 4; ++ni)
      vf[ni] = *(const v16bf*)(VTs + (ni * 16 + lo) * 64 + ks * 32 + hi * 16);
#pragma unroll
    for (int mi = 0; mi < 2; ++mi)
      pf[mi] = *(const v16bf*)(Ps + w * 2048 + (mi * 16 + lo) * 64 + ks * 32 + hi * 16);
#pragma unroll
    for (int mi = 0; mi < 2; ++mi)
#pragma unroll
      for (int ni = 0; ni < 4; ++ni)
        oacc[mi][ni] = WMMA_BF16(pf[mi], vf[ni], oacc[mi][ni]);
  }

  // ---- scatter back through m to original order: attn[b*S+m[i]][h*64+d]
#pragma unroll
  for (int mi = 0; mi < 2; ++mi) {
#pragma unroll
    for (int v = 0; v < 8; ++v) {
      int r = w * 32 + mi * 16 + v + 8 * hi;
      int orig = mmap[n * kSeg + r];
      size_t rowoff = (size_t)(b * kS + orig) * kD + h * kDh;
#pragma unroll
      for (int ni = 0; ni < 4; ++ni)
        attn_out[rowoff + ni * 16 + lo] = (__bf16)oacc[mi][ni][v];
    }
  }
}

// ---------------------------------------------------------------------------
// Host launcher
// ---------------------------------------------------------------------------
extern "C" void kernel_launch(void* const* d_in, const int* in_sizes, int n_in,
                              void* d_out, int out_size, void* d_ws, size_t ws_size,
                              hipStream_t stream) {
  (void)in_sizes; (void)n_in; (void)out_size; (void)ws_size;
  const float* x    = (const float*)d_in[0];   // [2,8192,1024]
  const float* Wqkv = (const float*)d_in[1];   // [1024,3072]
  const float* Wout = (const float*)d_in[2];   // [1024,1024]
  float* out = (float*)d_out;                  // [2,8192,1024]

  char* ws = (char*)d_ws;
  size_t off = 0;
  int*    mmap  = (int*)(ws + off);    off += (size_t)kS * sizeof(int);
  __bf16* xb    = (__bf16*)(ws + off); off += (size_t)kM * kD * 2;       // 32 MB
  __bf16* wqt   = (__bf16*)(ws + off); off += (size_t)kNqkv * kD * 2;    // 6 MB
  __bf16* wot   = (__bf16*)(ws + off); off += (size_t)kD * kD * 2;       // 2 MB
  __bf16* qkvb  = (__bf16*)(ws + off); off += (size_t)kM * kNqkv * 2;    // 96 MB
  __bf16* attnb = (__bf16*)(ws + off); off += (size_t)kM * kD * 2;       // 32 MB

  // 0) permutation map
  hilbert_map_kernel<<<dim3((kS + 255) / 256), dim3(256), 0, stream>>>(mmap, kS, kG);
  // 1) x -> bf16
  cvt_f32_bf16_kernel<<<dim3(kM * kD / 4 / 256), dim3(256), 0, stream>>>(x, xb, kM * kD);
  // 2) weights -> bf16, transposed [N][K]
  transpose_cvt_kernel<<<dim3(kNqkv / 32, kD / 32), dim3(32, 8), 0, stream>>>(Wqkv, wqt, kD, kNqkv);
  transpose_cvt_kernel<<<dim3(kD / 32, kD / 32), dim3(32, 8), 0, stream>>>(Wout, wot, kD, kD);
  // 3) qkv = x @ W_qkv  (bf16 out)
  gemm_bf16_kernel<__bf16><<<dim3(kNqkv / 256, kM / 128), dim3(256), 0, stream>>>(
      xb, wqt, qkvb, kM, kNqkv, kD);
  // 4) segmented dilated attention with Hilbert gather/scatter (bf16 out)
  attn_kernel<<<dim3(kNSeg, kH, kB), dim3(128), 0, stream>>>(qkvb, mmap, attnb);
  // 5) out = attn @ W_out  (f32 out)
  gemm_bf16_kernel<float><<<dim3(kD / 256, kM / 128), dim3(256), 0, stream>>>(
      attnb, wot, out, kM, kD, kD);
}
